// Mamba2Simple_1554778161717
// MI455X (gfx1250) — matrix operand order repaired and checked by reading, not compiled, not run on hardware
//
#include <hip/hip_runtime.h>
#include <hip/hip_bf16.h>

// ---------------- model constants ----------------
#define D_MODEL   1024
#define D_STATE   128
#define D_CONV    4
#define HEADDIM   64
#define CHUNK     256
#define D_INNER   2048
#define NHEADS    32
#define CONV_DIM  2304
#define D_IN_PROJ 4384
#define BATCH     2
#define SEQLEN    4096
#define NTOK      (BATCH * SEQLEN)      // 8192
#define NCHUNK    (SEQLEN / CHUNK)      // 16
#define NBC       (BATCH * NCHUNK)      // 32 chunk-blocks total
#define EPS       1e-5f

typedef unsigned int u32;
typedef __attribute__((ext_vector_type(16))) __bf16 v16bf;
typedef __attribute__((ext_vector_type(8)))  __bf16 v8bf;
typedef __attribute__((ext_vector_type(8)))  float  v8f;
typedef __attribute__((ext_vector_type(4)))  u32    u32x4;
typedef __attribute__((ext_vector_type(4)))  int    i32x4;
typedef __attribute__((ext_vector_type(8)))  int    i32x8;

// LDS tile leading dims. bf16 tiles: 40 (80B rows). f32 TDM-staged tile: 36
// (144B rows) — produced by TDM pad feature (4 DWORDs pad per 32 DWORDs).
#define LDT  40
#define LDTF 36

#if defined(__has_builtin)
#if __has_builtin(__builtin_amdgcn_tensor_load_to_lds)
#define HAVE_TDM 1
#endif
#endif

// ---------------- TDM: 2D f32 tile (rows x cols) global -> LDS -------------
// D# per CDNA5 ISA ch.8: group0 {count, lds_addr, global_addr, type=2},
// group1 {data_size=4B, pad_enable, pad 4DW per 32DW, dims, stride}.
#ifdef HAVE_TDM
__device__ inline void tdm_load_tile_f32(const float* gptr, u32 lds_bytes,
                                         u32 rows, u32 cols, u32 ld_elems) {
  unsigned long long ga = (unsigned long long)(uintptr_t)gptr;
  u32x4 g0;
  g0[0] = 1u;                                        // count=1 user descriptor
  g0[1] = lds_bytes;                                 // LDS byte address
  g0[2] = (u32)ga;                                   // global addr [31:0]
  g0[3] = ((u32)(ga >> 32) & 0x1FFFFFFu) | (2u << 30); // [56:32] | type=2
  i32x8 g1;
  // data_size=2(4B) | pad_enable | pad_interval=4(32DW) | pad_amount=3(4DW)
  g1[0] = (int)((2u << 16) | (1u << 20) | (4u << 22) | (3u << 25));
  u32 td0 = ld_elems;                                // tensor_dim0
  u32 td1 = rows;                                    // tensor_dim1
  g1[1] = (int)((td0 & 0xFFFFu) << 16);              // dim0[15:0] @ bits63:48
  g1[2] = (int)((td0 >> 16) | ((td1 & 0xFFFFu) << 16));
  g1[3] = (int)((td1 >> 16) | (cols << 16));         // tile_dim0 @ bits127:112
  g1[4] = (int)(rows & 0xFFFFu);                     // tile_dim1; tile_dim2=0
  g1[5] = (int)ld_elems;                             // dim0_stride[31:0]
  g1[6] = 0;                                         // dim0_stride[47:32]=0
  g1[7] = 0;
  i32x4 g2 = {0, 0, 0, 0};
  i32x4 g3 = {0, 0, 0, 0};
#if __clang_major__ >= 23
  i32x8 g4 = {0, 0, 0, 0, 0, 0, 0, 0};
  __builtin_amdgcn_tensor_load_to_lds(g0, g1, g2, g3, g4, 0);
#else
  __builtin_amdgcn_tensor_load_to_lds(g0, g1, g2, g3, 0);
#endif
}
#endif

// ---------------- WMMA fragment helpers (per CDNA5 ISA 7.12.2 layouts) -----
// A (16x32 bf16): lane(half,r) holds row r; elems 0..7 -> K = 8*half+0..7,
// elems 8..15 -> K = 16+8*half+0..7.
__device__ inline v16bf load_a_frag(const __bf16* tile, int row, int half) {
  const v8bf* p0 = (const v8bf*)(tile + row * LDT + 8 * half);
  const v8bf* p1 = (const v8bf*)(tile + row * LDT + 16 + 8 * half);
  v8bf a = *p0, b = *p1;
  v16bf r;
#pragma unroll
  for (int i = 0; i < 8; ++i) { r[i] = a[i]; r[i + 8] = b[i]; }
  return r;
}
// Same A-fragment mapping but from an f32 tile (TDM staged), cvt to bf16.
__device__ inline v16bf load_a_frag_f32(const float* tile, int row, int half) {
  const float4* p0 = (const float4*)(tile + row * LDTF + 8 * half);
  const float4* p1 = (const float4*)(tile + row * LDTF + 16 + 8 * half);
  float4 a0 = p0[0], a1 = p0[1], b0 = p1[0], b1 = p1[1];
  v16bf r;
  r[0] = (__bf16)a0.x; r[1] = (__bf16)a0.y; r[2] = (__bf16)a0.z; r[3] = (__bf16)a0.w;
  r[4] = (__bf16)a1.x; r[5] = (__bf16)a1.y; r[6] = (__bf16)a1.z; r[7] = (__bf16)a1.w;
  r[8]  = (__bf16)b0.x; r[9]  = (__bf16)b0.y; r[10] = (__bf16)b0.z; r[11] = (__bf16)b0.w;
  r[12] = (__bf16)b1.x; r[13] = (__bf16)b1.y; r[14] = (__bf16)b1.z; r[15] = (__bf16)b1.w;
  return r;
}
// B (32x16 bf16): lane(half,r) holds col r; elems 0..7 -> K = 8*half+0..7,
// elems 8..15 -> K = 16+8*half+0..7.
// Tile stored as [col][k] (k contiguous).
__device__ inline v16bf load_b_frag(const __bf16* tile, int col, int half) {
  const v8bf* p0 = (const v8bf*)(tile + col * LDT + 8 * half);
  const v8bf* p1 = (const v8bf*)(tile + col * LDT + 16 + 8 * half);
  v8bf a = *p0, b = *p1;
  v16bf r;
#pragma unroll
  for (int i = 0; i < 8; ++i) { r[i] = a[i]; r[i + 8] = b[i]; }
  return r;
}

__device__ inline v8f wmma_bf16(v16bf a, v16bf b, v8f c) {
  return __builtin_amdgcn_wmma_f32_16x16x32_bf16(false, a, false, b,
                                                 (short)0, c, false, false);
}

__device__ inline float silu_f(float x) { return x / (1.0f + __expf(-x)); }
__device__ inline float softplus_f(float x) {
  return (x > 20.0f) ? x : log1pf(__expf(x));
}

// =====================================================================
// Generic batched C = A * W^T   (A: MxK row-major, W: NxK row-major)
// Block tile 128(M) x 64(N), K-step 32, 256 threads = 8 waves.
// A tile staged f32 via Tensor Data Mover (dynamic LDS offset 0);
// W tile converted to bf16 on the VMEM path (overlaps the TDM DMA).
// Dynamic LDS layout: [0, 18432) f32 A tile (128x36), [18432, ...) bf16 W.
// =====================================================================
#define GEMM_SMEM (128 * LDTF * 4 + 64 * LDT * 2)

__global__ __launch_bounds__(256)
void k_gemm_xwt(const float* __restrict__ A, int lda, long sA,
                const float* __restrict__ W, int ldw, long sW,
                float* __restrict__ C, int ldc, long sC,
                int M, int N, int K) {
  extern __shared__ char smem[];
  float*  AsF = (float*)smem;                       // LDS byte offset 0
  __bf16* Ws  = (__bf16*)(smem + 128 * LDTF * 4);

  const float* Ab = A + (size_t)blockIdx.z * sA;
  const float* Wb = W + (size_t)blockIdx.z * sW;
  float*       Cb = C + (size_t)blockIdx.z * sC;

  const int t    = threadIdx.x;
  const int wave = t >> 5;
  const int lane = t & 31;
  const int half = lane >> 4;
  const int r    = lane & 15;
  const int n0   = blockIdx.x * 64;
  const int m0   = blockIdx.y * 128;

  v8f acc[4];
#pragma unroll
  for (int i = 0; i < 4; ++i) acc[i] = (v8f){0, 0, 0, 0, 0, 0, 0, 0};

  const int wrow = t >> 2;          // 0..63
  const int wkk  = (t & 3) * 8;     // 0,8,16,24
#ifndef HAVE_TDM
  const int arow = t >> 1;          // 0..127
  const int akk  = (t & 1) * 16;    // 0 / 16
#endif

  for (int kb = 0; kb < K; kb += 32) {
    __syncthreads();
#ifdef HAVE_TDM
    if (wave == 0)  // one TDM DMA per workgroup: 128x32 f32 tile, 36-f pitch
      tdm_load_tile_f32(Ab + (size_t)m0 * lda + kb, 0u, 128u, 32u, (u32)lda);
#else
    { // manual A tile fill (f32, 36-float pitch)
      const float4* ap = (const float4*)(Ab + (size_t)(m0 + arow) * lda + kb + akk);
#pragma unroll
      for (int q = 0; q < 4; ++q)
        *(float4*)&AsF[arow * LDTF + akk + q * 4] = ap[q];
    }
#endif
    { // W tile -> bf16 (guard N for in_proj's N=4384); overlaps the DMA
      const int gn = n0 + wrow;
      if (gn < N) {
        const float4* wp = (const float4*)(Wb + (size_t)gn * ldw + kb + wkk);
#pragma unroll
        for (int q = 0; q < 2; ++q) {
          float4 v = wp[q];
          __bf16* d = &Ws[wrow * LDT + wkk + q * 4];
          d[0] = (__bf16)v.x; d[1] = (__bf16)v.y; d[2] = (__bf16)v.z; d[3] = (__bf16)v.w;
        }
        if (kb + 32 < K)
          __builtin_prefetch(Wb + (size_t)gn * ldw + kb + 32 + wkk, 0, 1);
      } else {
#pragma unroll
        for (int q = 0; q < 8; ++q) Ws[wrow * LDT + wkk + q] = (__bf16)0.0f;
      }
    }
#ifdef HAVE_TDM
    if (wave == 0) __builtin_amdgcn_s_wait_tensorcnt(0);
#endif
    __syncthreads();

    v16bf af = load_a_frag_f32(AsF, wave * 16 + r, half);
#pragma unroll
    for (int nt = 0; nt < 4; ++nt) {
      v16bf bfr = load_b_frag(Ws, nt * 16 + r, half);
      acc[nt] = wmma_bf16(af, bfr, acc[nt]);
    }
  }

#pragma unroll
  for (int nt = 0; nt < 4; ++nt) {
    int col = n0 + nt * 16 + r;
    if (col < N) {
      float* cp = Cb + (size_t)(m0 + wave * 16 + 8 * half) * ldc + col;
#pragma unroll
      for (int v = 0; v < 8; ++v) cp[(size_t)v * ldc] = acc[nt][v];
    }
  }
}

// =====================================================================
// dt = softplus(dt_raw + dt_bias)        (NTOK x NHEADS)
// =====================================================================
__global__ void k_dt(const float* __restrict__ zxbcdt,
                     const float* __restrict__ dt_bias,
                     float* __restrict__ dtsp) {
  int idx = blockIdx.x * blockDim.x + threadIdx.x;
  if (idx >= NTOK * NHEADS) return;
  int bl = idx >> 5, h = idx & 31;
  float x = zxbcdt[(size_t)bl * D_IN_PROJ + (D_INNER + CONV_DIM) + h] + dt_bias[h];
  dtsp[idx] = softplus_f(x);
}

// =====================================================================
// causal depthwise conv1d + SiLU over xBC section of zxbcdt
// grid (CONV_DIM/256, NTOK)
// =====================================================================
__global__ void k_conv_silu(const float* __restrict__ zxbcdt,
                            const float* __restrict__ conv_w,
                            const float* __restrict__ conv_b,
                            float* __restrict__ xconv) {
  int ch = blockIdx.x * 256 + threadIdx.x;
  int bl = blockIdx.y;
  int l  = bl & (SEQLEN - 1);
  float acc = conv_b[ch];
#pragma unroll
  for (int k = 0; k < D_CONV; ++k) {
    int ll = l + k - (D_CONV - 1);
    if (ll >= 0)
      acc += zxbcdt[(size_t)(bl + k - (D_CONV - 1)) * D_IN_PROJ + D_INNER + ch] *
             conv_w[ch * D_CONV + k];
  }
  xconv[(size_t)bl * CONV_DIM + ch] = silu_f(acc);
}

// =====================================================================
// Per-chunk inclusive cumsum of dA = dt * A.   grid (NBC, NHEADS), 256 thr
// =====================================================================
__global__ void k_cumsum(const float* __restrict__ dtsp,
                         const float* __restrict__ A_log,
                         float* __restrict__ acum) {
  __shared__ float sc[CHUNK];
  int bc = blockIdx.x, h = blockIdx.y, s = threadIdx.x;
  int gl = bc * CHUNK + s;
  float Ah = -__expf(A_log[h]);
  sc[s] = dtsp[(size_t)gl * NHEADS + h] * Ah;
  __syncthreads();
  for (int off = 1; off < CHUNK; off <<= 1) {
    float v = (s >= off) ? sc[s - off] : 0.0f;
    __syncthreads();
    sc[s] += v;
    __syncthreads();
  }
  acum[(size_t)gl * NHEADS + h] = sc[s];
}

// =====================================================================
// chunk-final states: states[bc,h,p,n] = sum_s B[s,n]*decay[s]*x[s,p]*dt[s]
// WMMA GEMM  M=64(p) N=128(n) K=256(s).   grid (NBC, NHEADS), 256 thr
// =====================================================================
__global__ __launch_bounds__(256)
void k_states(const float* __restrict__ xconv,
              const float* __restrict__ dtsp,
              const float* __restrict__ acum,
              float* __restrict__ states) {
  __shared__ __bf16 Xs[64 * LDT];    // A tile: [p][s]
  __shared__ __bf16 Bs[128 * LDT];   // B tile: [n][s]
  __shared__ float  wdt[32];

  const int bc = blockIdx.x, h = blockIdx.y;
  const int gbase = bc * CHUNK;
  const int t = threadIdx.x, wave = t >> 5, lane = t & 31;
  const int half = lane >> 4, r = lane & 15;
  const int mt = wave & 3, ct0 = (wave >> 2) * 4;

  const float acl = acum[(size_t)(gbase + CHUNK - 1) * NHEADS + h];

  v8f acc[4];
#pragma unroll
  for (int i = 0; i < 4; ++i) acc[i] = (v8f){0, 0, 0, 0, 0, 0, 0, 0};

  for (int sb = 0; sb < CHUNK; sb += 32) {
    __syncthreads();
    if (t < 32) {
      size_t gi = (size_t)(gbase + sb + t) * NHEADS + h;
      wdt[t] = dtsp[gi] * __expf(acl - acum[gi]);
    }
    __syncthreads();
    { // Xs[p][s] = x * dt * decay
      int s = t & 31, p0 = (t >> 5) * 8;
      const float* xp = xconv + (size_t)(gbase + sb + s) * CONV_DIM + h * HEADDIM + p0;
      float w = wdt[s];
#pragma unroll
      for (int j = 0; j < 8; ++j) Xs[(p0 + j) * LDT + s] = (__bf16)(xp[j] * w);
    }
    { // Bs[n][s] = B-matrix
      int s = t & 31, nn0 = (t >> 5) * 16;
      const float* bp = xconv + (size_t)(gbase + sb + s) * CONV_DIM + D_INNER + nn0;
#pragma unroll
      for (int j = 0; j < 16; ++j) Bs[(nn0 + j) * LDT + s] = (__bf16)bp[j];
    }
    __syncthreads();

    v16bf af = load_a_frag(Xs, mt * 16 + r, half);
#pragma unroll
    for (int j = 0; j < 4; ++j) {
      v16bf bfr = load_b_frag(Bs, (ct0 + j) * 16 + r, half);
      acc[j] = wmma_bf16(af, bfr, acc[j]);
    }
  }

  float* sp = states + ((size_t)bc * NHEADS + h) * HEADDIM * D_STATE;
#pragma unroll
  for (int j = 0; j < 4; ++j) {
    int n = (ct0 + j) * 16 + r;
#pragma unroll
    for (int v = 0; v < 8; ++v) {
      int p = mt * 16 + 8 * half + v;
      sp[(size_t)p * D_STATE + n] = acc[j][v];
    }
  }
}

// =====================================================================
// sequential inter-chunk scan; rewrites states[] in place with the state
// ENTERING each chunk (prev).  one thread per (b,h,p,n)
// =====================================================================
__global__ void k_scan(float* __restrict__ states,
                       const float* __restrict__ acum) {
  int idx = blockIdx.x * blockDim.x + threadIdx.x;
  if (idx >= BATCH * NHEADS * HEADDIM * D_STATE) return;
  int n = idx & (D_STATE - 1);
  int p = (idx >> 7) & (HEADDIM - 1);
  int h = (idx >> 13) & (NHEADS - 1);
  int b = idx >> 18;
  float carry = 0.0f;
  for (int c = 0; c < NCHUNK; ++c) {
    int bc = b * NCHUNK + c;
    float cd = __expf(acum[(size_t)(bc * CHUNK + CHUNK - 1) * NHEADS + h]);
    size_t si = (((size_t)bc * NHEADS + h) * HEADDIM + p) * D_STATE + n;
    float st = states[si];
    states[si] = carry;            // prev: state entering chunk c
    carry = carry * cd + st;
  }
}

// =====================================================================
// Y = (G.*L) @ xdt  +  outdecay .* (C @ prev^T)  +  D*x
// per (bc,h): M=256(s), N=64(p).  grid (NBC, NHEADS), 256 thr = 8 waves.
// Wave w owns rows [w*32, w*32+32).  A fragments built in registers.
// =====================================================================
__global__ __launch_bounds__(256)
void k_output(const float* __restrict__ xconv,
              const float* __restrict__ dtsp,
              const float* __restrict__ acum,
              const float* __restrict__ G,
              const float* __restrict__ prev,   // states[] after k_scan
              const float* __restrict__ Dp,
              float* __restrict__ ybuf) {
  __shared__ float  AcS[CHUNK];
  __shared__ __bf16 Bs[64 * LDT];   // [p][k]

  const int bc = blockIdx.x, h = blockIdx.y;
  const int gbase = bc * CHUNK;
  const int t = threadIdx.x, wave = t >> 5, lane = t & 31;
  const int half = lane >> 4, r = lane & 15;
  const float* Gp = G + (size_t)bc * CHUNK * CHUNK;
  const float* pv = prev + ((size_t)bc * NHEADS + h) * HEADDIM * D_STATE;

  AcS[t] = acum[(size_t)(gbase + t) * NHEADS + h];
  __syncthreads();

  v8f acc[2][4];
#pragma unroll
  for (int a = 0; a < 2; ++a)
#pragma unroll
    for (int j = 0; j < 4; ++j) acc[a][j] = (v8f){0, 0, 0, 0, 0, 0, 0, 0};

  // ---- Phase 1: Y_diag, K-loop over z ----
  for (int zb = 0; zb < CHUNK; zb += 32) {
    __syncthreads();
    { // Bs[p][zz] = x[z,p]*dt[z]
      int zz = t & 31, p0 = (t >> 5) * 8;
      int gz = gbase + zb + zz;
      float dtv = dtsp[(size_t)gz * NHEADS + h];
      const float* xp = xconv + (size_t)gz * CONV_DIM + h * HEADDIM + p0;
#pragma unroll
      for (int j = 0; j < 8; ++j) Bs[(p0 + j) * LDT + zz] = (__bf16)(xp[j] * dtv);
    }
    __syncthreads();

    v16bf af[2];
#pragma unroll
    for (int rt = 0; rt < 2; ++rt) {
      int s = wave * 32 + rt * 16 + r;
      float acs = AcS[s];
#pragma unroll
      for (int i = 0; i < 16; ++i) {
        int z = zb + ((i & 7) + ((i >> 3) << 4) + (half << 3));
        float val = (z <= s) ? Gp[(size_t)s * CHUNK + z] * __expf(acs - AcS[z]) : 0.0f;
        af[rt][i] = (__bf16)val;
      }
    }
#pragma unroll
    for (int rt = 0; rt < 2; ++rt)
#pragma unroll
      for (int j = 0; j < 4; ++j) {
        v16bf bfr = load_b_frag(Bs, j * 16 + r, half);
        acc[rt][j] = wmma_bf16(af[rt], bfr, acc[rt][j]);
      }
  }

  // ---- Phase 2: Y_off, K-loop over n (state dim) ----
  for (int nb = 0; nb < D_STATE; nb += 32) {
    __syncthreads();
    { // Bs[p][kk] = prev[p, nb+kk]
      int p = t & 63, k0 = (t >> 6) * 8;
      const float* pp = pv + (size_t)p * D_STATE + nb + k0;
#pragma unroll
      for (int j = 0; j < 8; ++j) Bs[p * LDT + k0 + j] = (__bf16)pp[j];
    }
    __syncthreads();

    v16bf af[2];
#pragma unroll
    for (int rt = 0; rt < 2; ++rt) {
      int s = wave * 32 + rt * 16 + r;
      float od = __expf(AcS[s]);
      const float* cp = xconv + (size_t)(gbase + s) * CONV_DIM + (D_INNER + D_STATE);
#pragma unroll
      for (int i = 0; i < 16; ++i) {
        int n = nb + ((i & 7) + ((i >> 3) << 4) + (half << 3));
        af[rt][i] = (__bf16)(cp[n] * od);
      }
    }
#pragma unroll
    for (int rt = 0; rt < 2; ++rt)
#pragma unroll
      for (int j = 0; j < 4; ++j) {
        v16bf bfr = load_b_frag(Bs, j * 16 + r, half);
        acc[rt][j] = wmma_bf16(af[rt], bfr, acc[rt][j]);
      }
  }

  // ---- epilogue: + D*x, store y (token-major, d_inner channel order) ----
  const float Dph = Dp[h];
#pragma unroll
  for (int rt = 0; rt < 2; ++rt)
#pragma unroll
    for (int j = 0; j < 4; ++j) {
      int p = j * 16 + r;
#pragma unroll
      for (int v = 0; v < 8; ++v) {
        int s = wave * 32 + rt * 16 + 8 * half + v;
        size_t gi = (size_t)(gbase + s);
        float xv = xconv[gi * CONV_DIM + h * HEADDIM + p];
        ybuf[gi * D_INNER + h * HEADDIM + p] = acc[rt][j][v] + Dph * xv;
      }
    }
}

// =====================================================================
// gated SiLU + RMSNorm (in place on ybuf).  one block per token.
// =====================================================================
__global__ __launch_bounds__(256)
void k_gate_rmsnorm(float* __restrict__ ybuf,
                    const float* __restrict__ zxbcdt,
                    const float* __restrict__ norm_w) {
  __shared__ float red[256];
  const int tok = blockIdx.x, t = threadIdx.x;
  float g[8];
  float ss = 0.0f;
#pragma unroll
  for (int k = 0; k < 8; ++k) {
    int ch = k * 256 + t;
    float yv = ybuf[(size_t)tok * D_INNER + ch];
    float zv = zxbcdt[(size_t)tok * D_IN_PROJ + ch];
    g[k] = yv * silu_f(zv);
    ss += g[k] * g[k];
  }
  red[t] = ss;
  __syncthreads();
  for (int off = 128; off > 0; off >>= 1) {
    if (t < off) red[t] += red[t + off];
    __syncthreads();
  }
  float scale = rsqrtf(red[0] / (float)D_INNER + EPS);
#pragma unroll
  for (int k = 0; k < 8; ++k) {
    int ch = k * 256 + t;
    ybuf[(size_t)tok * D_INNER + ch] = g[k] * scale * norm_w[ch];
  }
}

// =====================================================================
extern "C" void kernel_launch(void* const* d_in, const int* in_sizes, int n_in,
                              void* d_out, int out_size, void* d_ws, size_t ws_size,
                              hipStream_t stream) {
  const float* u       = (const float*)d_in[0];
  const float* in_w    = (const float*)d_in[1];
  const float* conv_w  = (const float*)d_in[2];
  const float* conv_b  = (const float*)d_in[3];
  const float* dt_bias = (const float*)d_in[4];
  const float* A_log   = (const float*)d_in[5];
  const float* Dp      = (const float*)d_in[6];
  const float* norm_w  = (const float*)d_in[7];
  const float* out_w   = (const float*)d_in[8];
  float* out = (float*)d_out;

  float* ws = (float*)d_ws;
  float* zxbcdt = ws;                                   // 8192*4384
  float* xconv  = zxbcdt + (size_t)NTOK * D_IN_PROJ;    // 8192*2304
  float* dtsp   = xconv  + (size_t)NTOK * CONV_DIM;     // 8192*32
  float* acum   = dtsp   + (size_t)NTOK * NHEADS;       // 8192*32
  float* G      = acum   + (size_t)NTOK * NHEADS;       // 32*256*256
  float* states = G      + (size_t)NBC * CHUNK * CHUNK; // 32*32*64*128
  float* ybuf   = states + (size_t)NBC * NHEADS * HEADDIM * D_STATE; // 8192*2048

  dim3 blk(256);

  // 1) in_proj: zxbcdt = u @ in_w^T   (M=8192, N=4384, K=1024)
  k_gemm_xwt<<<dim3((D_IN_PROJ + 63) / 64, NTOK / 128, 1), blk, GEMM_SMEM, stream>>>(
      u, D_MODEL, 0, in_w, D_MODEL, 0, zxbcdt, D_IN_PROJ, 0,
      NTOK, D_IN_PROJ, D_MODEL);

  // 2) dt softplus
  k_dt<<<dim3((NTOK * NHEADS + 255) / 256), blk, 0, stream>>>(zxbcdt, dt_bias, dtsp);

  // 3) causal conv + SiLU
  k_conv_silu<<<dim3(CONV_DIM / 256, NTOK), blk, 0, stream>>>(zxbcdt, conv_w, conv_b, xconv);

  // 4) per-chunk cumsum of dt*A
  k_cumsum<<<dim3(NBC, NHEADS), blk, 0, stream>>>(dtsp, A_log, acum);

  // 5) G[bc] = C_chunk @ B_chunk^T   (batched, M=N=256, K=128)
  k_gemm_xwt<<<dim3(CHUNK / 64, CHUNK / 128, NBC), blk, GEMM_SMEM, stream>>>(
      xconv + (D_INNER + D_STATE), CONV_DIM, (long)CHUNK * CONV_DIM,
      xconv + D_INNER,            CONV_DIM, (long)CHUNK * CONV_DIM,
      G, CHUNK, (long)CHUNK * CHUNK,
      CHUNK, CHUNK, D_STATE);

  // 6) chunk-final states (WMMA)
  k_states<<<dim3(NBC, NHEADS), blk, 0, stream>>>(xconv, dtsp, acum, states);

  // 7) inter-chunk scan (states -> prev, in place)
  k_scan<<<dim3((BATCH * NHEADS * HEADDIM * D_STATE) / 256), blk, 0, stream>>>(states, acum);

  // 8) chunk output: Y_diag + Y_off + D*x (WMMA)
  k_output<<<dim3(NBC, NHEADS), blk, 0, stream>>>(xconv, dtsp, acum, G, states, Dp, ybuf);

  // 9) gate + RMSNorm (in place)
  k_gate_rmsnorm<<<dim3(NTOK), blk, 0, stream>>>(ybuf, zxbcdt, norm_w);

  // 10) out_proj: out = y @ out_w^T   (M=8192, N=1024, K=2048)
  k_gemm_xwt<<<dim3(D_MODEL / 64, NTOK / 128, 1), blk, GEMM_SMEM, stream>>>(
      ybuf, D_INNER, 0, out_w, D_INNER, 0, out, D_MODEL, 0,
      NTOK, D_MODEL, D_INNER);
}
